// BiMambaBlock_43250320670965
// MI455X (gfx1250) — matrix operand order repaired and checked by reading, not compiled, not run on hardware
//
#include <hip/hip_runtime.h>
#include <hip/hip_bf16.h>
#include <math.h>

// ---------------- problem constants ----------------
#define NB      2
#define SEQ     2048
#define DMODEL  1024
#define DI      2048      // d_inner
#define DSTATE  16
#define DTRANK  64
#define NPROJ   96        // DT_RANK + 2*D_STATE
#define MTOT    (NB*SEQ)  // 4096 rows of activations

// ---------------- types ----------------
typedef __bf16 bf16;
typedef bf16         v16bf __attribute__((ext_vector_type(16)));
typedef float        v8f   __attribute__((ext_vector_type(8)));
typedef unsigned int u32x4 __attribute__((ext_vector_type(4)));

union Frag32B { u32x4 q[2]; v16bf v; bf16 h[16]; };
union Frag16B { u32x4 q;            bf16 h[8];  };

// =====================================================================
// GEMM: C[M][N] = A[M][K] * W[N][K]^T   (fp32 in/out, bf16 WMMA compute)
// BM=128, BN=64, BK=32; 256 threads = 8 waves (4x2), wave tile 32x32.
// =====================================================================
#define BM 128
#define BN 64
#define BK 32
#define LDSK (BK + 8)   // 40 bf16 = 80B row stride (16B aligned, conflict-free)

__global__ __launch_bounds__(256)
void gemm_bf16wmma(const float* __restrict__ A, const float* __restrict__ W,
                   float* __restrict__ C, int M, int N, int K,
                   int lda, int ldw, int ldc)
{
    __shared__ bf16 As[BM][LDSK];
    __shared__ bf16 Ws[BN][LDSK];

    const int tid  = threadIdx.x;
    const int lane = tid & 31;
    const int wave = tid >> 5;
    const int wm   = wave >> 1;          // 0..3  (row of 32)
    const int wn   = wave & 1;           // 0..1  (col of 32)
    const int m0   = blockIdx.y * BM;
    const int n0   = blockIdx.x * BN;

    // staging assignments
    const int arow  = tid >> 1;          // 0..127
    const int akoff = (tid & 1) * 16;    // 0 / 16
    const int wrow  = tid >> 2;          // 0..63
    const int wkoff = (tid & 3) * 8;     // 0 / 8 / 16 / 24

    // fragment lane decode
    const int mrow   = lane & 15;
    const int khalfA = (lane >> 4) * 16;  // A: 0 or 16
    const int kbaseB = (lane >> 4) * 16;  // B: 0 or 16

    v8f acc[2][2] = {};

    for (int k0 = 0; k0 < K; k0 += BK) {
        // ---- stage A tile (always in-bounds: M multiple of BM here) ----
        {
            const float* src = A + (size_t)(m0 + arow) * lda + k0 + akoff;
            Frag32B t;
            #pragma unroll
            for (int i = 0; i < 16; ++i) t.h[i] = (bf16)src[i];
            u32x4* dst = (u32x4*)&As[arow][akoff];
            dst[0] = t.q[0];
            dst[1] = t.q[1];
        }
        // ---- stage W tile (guard N: e.g. x_proj has N=96) ----
        {
            const int gn = n0 + wrow;
            Frag16B t;
            if (gn < N) {
                const float* src = W + (size_t)gn * ldw + k0 + wkoff;
                #pragma unroll
                for (int i = 0; i < 8; ++i) t.h[i] = (bf16)src[i];
            } else {
                #pragma unroll
                for (int i = 0; i < 8; ++i) t.h[i] = (bf16)0.0f;
            }
            *(u32x4*)&Ws[wrow][wkoff] = t.q;
        }
        __syncthreads();

        // ---- build fragments from LDS (contiguous k = 16*(lane/16) + s, same for A and W) ----
        v16bf afrag[2], wfrag[2];
        #pragma unroll
        for (int i = 0; i < 2; ++i) {
            const bf16* ap = &As[wm * 32 + i * 16 + mrow][0];
            Frag32B f;
            f.q[0] = *(const u32x4*)(ap + khalfA);        // K = base+0..7
            f.q[1] = *(const u32x4*)(ap + 8 + khalfA);    // K = base+8..15
            afrag[i] = f.v;
        }
        #pragma unroll
        for (int j = 0; j < 2; ++j) {
            const bf16* wp = &Ws[wn * 32 + j * 16 + mrow][kbaseB];
            Frag32B f;
            f.q[0] = *(const u32x4*)(wp);                 // K = kbase+0..7
            f.q[1] = *(const u32x4*)(wp + 8);             // K = kbase+8..15
            wfrag[j] = f.v;
        }

        #pragma unroll
        for (int i = 0; i < 2; ++i)
            #pragma unroll
            for (int j = 0; j < 2; ++j)
                acc[i][j] = __builtin_amdgcn_wmma_f32_16x16x32_bf16(
                    false, afrag[i], false, wfrag[j],
                    (short)0, acc[i][j], false, false);
        __syncthreads();
    }

    // ---- epilogue: C/D layout VGPR v -> row v + 8*(lane>=16), col lane%16 ----
    const int rhalf = (lane >> 4) * 8;
    #pragma unroll
    for (int i = 0; i < 2; ++i) {
        #pragma unroll
        for (int j = 0; j < 2; ++j) {
            const int gn = n0 + wn * 32 + j * 16 + mrow;
            if (gn >= N) continue;
            #pragma unroll
            for (int v = 0; v < 8; ++v) {
                const int gm = m0 + wm * 32 + i * 16 + rhalf + v;
                C[(size_t)gm * ldc + gn] = acc[i][j][v];
            }
        }
    }
}

// =====================================================================
// Depthwise causal conv (K=4) + SiLU.  flip!=0 reads the time-reversed
// sequence (bwd branch).  Input has leading-dim ldx (x_in lives inside xz).
// =====================================================================
__global__ __launch_bounds__(256)
void conv_silu(const float* __restrict__ xin, const float* __restrict__ w,
               const float* __restrict__ bias, float* __restrict__ out,
               int ldx, int flip)
{
    const int d = blockIdx.x * 256 + threadIdx.x;   // 0..DI-1
    const int t = blockIdx.y;                       // 0..SEQ-1
    const int b = blockIdx.z;                       // 0..NB-1

    float acc = bias[d];
    #pragma unroll
    for (int j = 0; j < 4; ++j) {
        const int tt = t - 3 + j;                   // causal (left pad 3)
        if (tt >= 0) {
            const int src = flip ? (SEQ - 1 - tt) : tt;
            acc = fmaf(w[d * 4 + j], xin[((size_t)b * SEQ + src) * ldx + d], acc);
        }
    }
    const float s = acc / (1.0f + __expf(-acc));    // silu
    out[((size_t)b * SEQ + t) * DI + d] = s;
}

// =====================================================================
// dt = softplus(dt_pre + dt_b)   (in place)
// =====================================================================
__global__ __launch_bounds__(256)
void softplus_bias(float* __restrict__ dt, const float* __restrict__ bias, int total)
{
    const int i = blockIdx.x * 256 + threadIdx.x;
    if (i >= total) return;
    const float x = dt[i] + bias[i & (DI - 1)];
    dt[i] = (x > 20.0f) ? x : log1pf(__expf(x));
}

// =====================================================================
// Selective scan.  One lane per (channel d, state n): 16 lanes/channel,
// serial over t with per-step shfl_xor reduction of h*C within the
// 16-lane group.  B/C/x/dt staged into LDS in 32-step chunks.
// Output y * silu(z) written at flipped position for the bwd branch;
// accumulate!=0 does += (fwd scan ran first and initialized).
// =====================================================================
#define CHB 16   // channels per 256-thread block

__global__ __launch_bounds__(256)
void scan_kernel(const float* __restrict__ xc,   // [NB][SEQ][DI]
                 const float* __restrict__ dt,   // [NB][SEQ][DI]
                 const float* __restrict__ proj, // [NB][SEQ][NPROJ] (B@64, C@80)
                 const float* __restrict__ A_log,// [DI][DSTATE]
                 const float* __restrict__ Dp,   // [DI]
                 const float* __restrict__ zbuf, // z, leading dim ldz
                 float* __restrict__ ysum,       // leading dim ldy
                 int ldz, int ldy, int flip, int accumulate)
{
    const int tid  = threadIdx.x;
    const int n    = tid & 15;            // state index
    const int cloc = tid >> 4;            // local channel 0..15
    const int d    = blockIdx.x * CHB + cloc;
    const int b    = blockIdx.y;

    __shared__ float sB[32][DSTATE];
    __shared__ float sC[32][DSTATE];
    __shared__ float sx[32][CHB];
    __shared__ float sdt[32][CHB];

    const float An = -__expf(A_log[d * DSTATE + n]);
    const float Dd = Dp[d];
    const size_t baseRow = (size_t)b * SEQ;

    float h = 0.0f;

    for (int t0 = 0; t0 < SEQ; t0 += 32) {
        __syncthreads();
        // B/C: 32 steps x 32 values
        for (int i = tid; i < 32 * 32; i += 256) {
            const int tt = i >> 5, c = i & 31;
            const float v = proj[(baseRow + t0 + tt) * NPROJ + DTRANK + c];
            if (c < DSTATE) sB[tt][c] = v; else sC[tt][c - DSTATE] = v;
        }
        // x, dt: 32 steps x 16 channels
        for (int i = tid; i < 32 * CHB; i += 256) {
            const int tt = i >> 4, cc = i & (CHB - 1);
            const size_t idx = (baseRow + t0 + tt) * DI + (size_t)blockIdx.x * CHB + cc;
            sx[tt][cc]  = xc[idx];
            sdt[tt][cc] = dt[idx];
        }
        __syncthreads();

        for (int k = 0; k < 32; ++k) {
            const float x   = sx[k][cloc];
            const float dtv = sdt[k][cloc];
            const float dA  = __expf(dtv * An);
            h = dA * h + (dtv * x) * sB[k][n];      // serial chain: one FMA
            float p = h * sC[k][n];
            p += __shfl_xor(p, 8, 16);
            p += __shfl_xor(p, 4, 16);
            p += __shfl_xor(p, 2, 16);
            p += __shfl_xor(p, 1, 16);
            if (n == 0) {
                const int t = t0 + k;
                const size_t tout = flip ? (size_t)(SEQ - 1 - t) : (size_t)t;
                const float z  = zbuf[(baseRow + tout) * ldz + d];
                const float y  = p + Dd * x;
                const float val = y * (z / (1.0f + __expf(-z)));
                float* op = &ysum[(baseRow + tout) * ldy + d];
                if (accumulate) *op += val; else *op = val;
            }
        }
    }
}

// =====================================================================
// launch
// =====================================================================
extern "C" void kernel_launch(void* const* d_in, const int* in_sizes, int n_in,
                              void* d_out, int out_size, void* d_ws, size_t ws_size,
                              hipStream_t stream)
{
    (void)in_sizes; (void)n_in; (void)out_size; (void)ws_size;

    const float* u     = (const float*)d_in[0];
    const float* inw   = (const float*)d_in[1];   // [2*DI][DMODEL]
    const float* outw  = (const float*)d_in[2];   // [DMODEL][DI]
    const float* fcw   = (const float*)d_in[3];
    const float* fcb   = (const float*)d_in[4];
    const float* fxpw  = (const float*)d_in[5];
    const float* fdtw  = (const float*)d_in[6];
    const float* fdtb  = (const float*)d_in[7];
    const float* fAlog = (const float*)d_in[8];
    const float* fD    = (const float*)d_in[9];
    const float* bcw   = (const float*)d_in[10];
    const float* bcb   = (const float*)d_in[11];
    const float* bxpw  = (const float*)d_in[12];
    const float* bdtw  = (const float*)d_in[13];
    const float* bdtb  = (const float*)d_in[14];
    const float* bAlog = (const float*)d_in[15];
    const float* bD    = (const float*)d_in[16];

    char* ws = (char*)d_ws;
    float* xz     = (float*)(ws);                          // [4096][4096] 64MB
    float* xc_f   = (float*)(ws + ((size_t)64  << 20));    // [4096][2048] 32MB
    float* xc_b   = (float*)(ws + ((size_t)96  << 20));    // 32MB
    float* proj_f = (float*)(ws + ((size_t)128 << 20));    // [4096][96]   2MB slot
    float* proj_b = (float*)(ws + ((size_t)130 << 20));    // 2MB slot
    float* dt_f   = (float*)(ws + ((size_t)132 << 20));    // [4096][2048] 32MB
    float* dt_b   = (float*)(ws + ((size_t)164 << 20));    // 32MB  (196MB total)

    float* zptr = xz + DI;   // z = xz cols 2048..4095, ld 4096
    float* yptr = xz;        // y_fwd+y_bwd reuses dead x_in half, ld 4096

    const dim3 blk(256);

    // 1) xz = u @ in_proj_w.T : [4096 x 4096], K=1024
    gemm_bf16wmma<<<dim3((2 * DI) / BN, MTOT / BM), blk, 0, stream>>>(
        u, inw, xz, MTOT, 2 * DI, DMODEL, DMODEL, DMODEL, 2 * DI);

    // 2-3) depthwise causal conv + silu (fwd normal, bwd on flipped seq)
    conv_silu<<<dim3(DI / 256, SEQ, NB), blk, 0, stream>>>(xz, fcw, fcb, xc_f, 2 * DI, 0);
    conv_silu<<<dim3(DI / 256, SEQ, NB), blk, 0, stream>>>(xz, bcw, bcb, xc_b, 2 * DI, 1);

    // 4-5) proj = xc @ x_proj_w.T : [4096 x 96], K=2048
    gemm_bf16wmma<<<dim3((NPROJ + BN - 1) / BN, MTOT / BM), blk, 0, stream>>>(
        xc_f, fxpw, proj_f, MTOT, NPROJ, DI, DI, DI, NPROJ);
    gemm_bf16wmma<<<dim3((NPROJ + BN - 1) / BN, MTOT / BM), blk, 0, stream>>>(
        xc_b, bxpw, proj_b, MTOT, NPROJ, DI, DI, DI, NPROJ);

    // 6-7) dt_pre = proj[:, :64] @ dt_w.T : [4096 x 2048], K=64 (lda=96 slice)
    gemm_bf16wmma<<<dim3(DI / BN, MTOT / BM), blk, 0, stream>>>(
        proj_f, fdtw, dt_f, MTOT, DI, DTRANK, NPROJ, DTRANK, DI);
    gemm_bf16wmma<<<dim3(DI / BN, MTOT / BM), blk, 0, stream>>>(
        proj_b, bdtw, dt_b, MTOT, DI, DTRANK, NPROJ, DTRANK, DI);

    // 8-9) dt = softplus(dt_pre + dt_b)
    const int dtTotal = MTOT * DI;
    softplus_bias<<<dim3(dtTotal / 256), blk, 0, stream>>>(dt_f, fdtb, dtTotal);
    softplus_bias<<<dim3(dtTotal / 256), blk, 0, stream>>>(dt_b, bdtb, dtTotal);

    // 10-11) selective scans; fwd writes, bwd accumulates at flipped t
    scan_kernel<<<dim3(DI / CHB, NB), blk, 0, stream>>>(
        xc_f, dt_f, proj_f, fAlog, fD, zptr, yptr, 2 * DI, 2 * DI, 0, 0);
    scan_kernel<<<dim3(DI / CHB, NB), blk, 0, stream>>>(
        xc_b, dt_b, proj_b, bAlog, bD, zptr, yptr, 2 * DI, 2 * DI, 1, 1);

    // 12) out = ysum @ out_proj_w.T : [4096 x 1024], K=2048 (lda=4096 slice)
    gemm_bf16wmma<<<dim3(DMODEL / BN, MTOT / BM), blk, 0, stream>>>(
        yptr, outw, (float*)d_out, MTOT, DMODEL, DI, 2 * DI, DI, DMODEL);
}
